// Seq2SeqModel_14920716387068
// MI455X (gfx1250) — matrix operand order repaired and checked by reading, not compiled, not run on hardware
//
#include <hip/hip_runtime.h>
#include <hip/hip_bf16.h>

// ---------------------------------------------------------------------------
// Types for CDNA5 WMMA (wave32, v_wmma_f32_16x16x32_bf16)
// ---------------------------------------------------------------------------
typedef __bf16 bf16_t;
typedef __attribute__((ext_vector_type(16))) __bf16 v16bf;
typedef __attribute__((ext_vector_type(8)))  __bf16 v8bf;
typedef __attribute__((ext_vector_type(8)))  float  v8f;

union ABfrag { v16bf v; v8bf h[2]; };

// Problem constants (match reference)
#define BSZ   128
#define SLEN  256
#define TLEN  256
#define EMB   256
#define ENCH  256   // per-direction hidden
#define DECH  512
#define DOF_  4
#define LB    4

// ---------------------------------------------------------------------------
// Generic WMMA GEMM:  C[M,N] = A[M,K](bf16, row-major, lda) @ W[N,K]^T (+bias)
// K must be a multiple of 32; M,N multiples of 16; N multiple of 64 per grid.
// Block = 128 threads = 4 waves; each wave computes one 16x16 tile of C.
// grid.x = N/64, grid.y = M/16.
// ---------------------------------------------------------------------------
__global__ __launch_bounds__(128)
void wmma_gemm_bf16(const bf16_t* __restrict__ A, int lda,
                    const bf16_t* __restrict__ Wt, int ldw,
                    const float* __restrict__ bias,
                    float* __restrict__ C, int ldc, int K) {
    const int wave = threadIdx.x >> 5;
    const int lane = threadIdx.x & 31;
    const int half = lane >> 4;
    const int r    = lane & 15;
    const int n0 = (blockIdx.x * 4 + wave) * 16;
    const int m0 = blockIdx.y * 16;

    // A fragment: row m0+r, K chunks [half*8 .. +7] and [16+half*8 .. +7]
    const bf16_t* pa = A  + (size_t)(m0 + r) * lda + half * 8;
    // B fragment: column n0+r of B == row n0+r of W; K chunks [half*8 .. +7] and [16+half*8 .. +7]
    const bf16_t* pw = Wt + (size_t)(n0 + r) * ldw + half * 8;

    v8f acc = {0.f, 0.f, 0.f, 0.f, 0.f, 0.f, 0.f, 0.f};
    for (int k = 0; k < K; k += 32) {
        ABfrag a;
        a.h[0] = *(const v8bf*)(pa + k);
        a.h[1] = *(const v8bf*)(pa + k + 16);
        ABfrag b;
        b.h[0] = *(const v8bf*)(pw + k);
        b.h[1] = *(const v8bf*)(pw + k + 16);
        acc = __builtin_amdgcn_wmma_f32_16x16x32_bf16(
                  false, a.v, false, b.v, (short)0, acc, false, false);
    }
    const float bv = bias ? bias[n0 + r] : 0.f;
    float* pc = C + (size_t)(m0 + half * 8) * ldc + n0 + r;
#pragma unroll
    for (int v = 0; v < 8; ++v)
        pc[(size_t)v * ldc] = acc[v] + bv;
}

// ---------------------------------------------------------------------------
// Elementwise helpers
// ---------------------------------------------------------------------------
__global__ void f32_to_bf16_kernel(const float* __restrict__ src,
                                   bf16_t* __restrict__ dst, int n) {
    int i = blockIdx.x * blockDim.x + threadIdx.x;
    if (i < n) dst[i] = (bf16_t)src[i];
}

// dst[r, 0:k1] = W1[r,:], dst[r, k1:k1+k2] = W2[r,:], rest zero.  ldd >= k1+k2
__global__ void pack2_bf16_kernel(const float* __restrict__ W1, int k1,
                                  const float* __restrict__ W2, int k2,
                                  bf16_t* __restrict__ dst, int ldd, int rows) {
    int i = blockIdx.x * blockDim.x + threadIdx.x;
    if (i >= rows * ldd) return;
    int r = i / ldd, c = i % ldd;
    float v = 0.f;
    if (c < k1)            v = W1[(size_t)r * k1 + c];
    else if (c < k1 + k2)  v = W2[(size_t)r * k2 + (c - k1)];
    dst[i] = (bf16_t)v;
}

// emb_bf [S,B,E] (time-major) from embedding table gather
__global__ void embed_kernel(const int* __restrict__ in_seq,
                             const float* __restrict__ table,
                             bf16_t* __restrict__ out) {
    int i = blockIdx.x * blockDim.x + threadIdx.x;       // S*B*E
    if (i >= SLEN * BSZ * EMB) return;
    int e = i & (EMB - 1);
    int b = (i >> 8) & (BSZ - 1);
    int s = i >> 15;
    int tok = in_seq[b * SLEN + s];
    out[i] = (bf16_t)table[(size_t)tok * EMB + e];
}

// win_bf [T,B,LB*DOF]: win[b,t,k*DOF+d] = (t+k<LB) ? 0 : tgt[b,t+k-LB,d]
__global__ void win_kernel(const float* __restrict__ tgt,
                           bf16_t* __restrict__ win) {
    int i = blockIdx.x * blockDim.x + threadIdx.x;       // T*B*16
    if (i >= TLEN * BSZ * LB * DOF_) return;
    int d = i & 3;
    int k = (i >> 2) & 3;
    int b = (i >> 4) & (BSZ - 1);
    int t = i >> 11;
    int st = t + k - LB;
    float v = (st >= 0) ? tgt[((size_t)b * TLEN + st) * DOF_ + d] : 0.f;
    win[i] = (bf16_t)v;
}

// copy emb slice t (fwd) and tb (bwd) into x-concat buffers (cols 0..EMB-1)
__global__ void enc_setx_kernel(const bf16_t* __restrict__ emb, int tf, int tb,
                                bf16_t* __restrict__ xf, bf16_t* __restrict__ xb) {
    int i = blockIdx.x * blockDim.x + threadIdx.x;       // B*E
    if (i >= BSZ * EMB) return;
    int b = i >> 8, e = i & (EMB - 1);
    xf[b * (EMB + ENCH) + e] = emb[((size_t)tf * BSZ + b) * EMB + e];
    xb[b * (EMB + ENCH) + e] = emb[((size_t)tb * BSZ + b) * EMB + e];
}

__global__ void dec_setwin_kernel(const bf16_t* __restrict__ win, int t,
                                  bf16_t* __restrict__ x) {   // ld = 1056
    int i = blockIdx.x * blockDim.x + threadIdx.x;       // B*16
    if (i >= BSZ * LB * DOF_) return;
    int b = i >> 4, k = i & 15;
    x[b * 1056 + k] = win[((size_t)t * BSZ + b) * 16 + k];
}

// LSTM cell epilogue: gates [B,4H] (i,f,g,o) -> c,h; h written as bf16 to two
// destinations (next-step x-concat, and enc_out / hc_cat), optionally f32.
__global__ void lstm_cell_kernel(const float* __restrict__ gates,
                                 float* __restrict__ c,
                                 bf16_t* __restrict__ dst1, int ld1,
                                 bf16_t* __restrict__ dst2, long ld2,
                                 float* __restrict__ hf32,
                                 int B, int H) {
    int idx = blockIdx.x * blockDim.x + threadIdx.x;
    if (idx >= B * H) return;
    int b = idx / H, j = idx % H;
    const float* g = gates + (size_t)b * 4 * H;
    float gi = g[j], gf = g[H + j], gg = g[2 * H + j], go = g[3 * H + j];
    float si = 1.f / (1.f + __expf(-gi));
    float sf = 1.f / (1.f + __expf(-gf));
    float so = 1.f / (1.f + __expf(-go));
    float cn = sf * c[idx] + si * tanhf(gg);
    c[idx] = cn;
    float hn = so * tanhf(cn);
    bf16_t hb = (bf16_t)hn;
    dst1[(size_t)b * ld1 + j] = hb;
    dst2[(size_t)b * ld2 + j] = hb;
    if (hf32) hf32[idx] = hn;
}

// scores[b,s] = h[b,:] . proj[b,s,:]; mask (in_seq==0) -> -inf; softmax over s.
// One block per b, thread = s.
__global__ __launch_bounds__(256)
void attn_scores_kernel(const float* __restrict__ h,
                        const float* __restrict__ proj,
                        const int* __restrict__ in_seq,
                        float* __restrict__ aw,
                        float* __restrict__ attn_out, int t) {
    __shared__ float sh[SLEN];
    int b = blockIdx.x, s = threadIdx.x;
    const float* hb = h + (size_t)b * DECH;
    const float* pb = proj + ((size_t)b * SLEN + s) * DECH;
    float acc = 0.f;
    for (int d = 0; d < DECH; ++d) acc += hb[d] * pb[d];
    bool mask = (in_seq[b * SLEN + s] == 0);
    if (mask) acc = -__builtin_inff();
    sh[s] = acc;
    __syncthreads();
    for (int off = 128; off > 0; off >>= 1) {
        if (s < off) sh[s] = fmaxf(sh[s], sh[s + off]);
        __syncthreads();
    }
    float m = sh[0];
    __syncthreads();
    float e = mask ? 0.f : __expf(acc - m);
    sh[s] = e;
    __syncthreads();
    for (int off = 128; off > 0; off >>= 1) {
        if (s < off) sh[s] += sh[s + off];
        __syncthreads();
    }
    float w = e / sh[0];
    aw[b * SLEN + s] = w;
    attn_out[((size_t)b * TLEN + t) * SLEN + s] = w;
}

// ctx[b,e] = sum_s aw[b,s] * enc_out[b,s,e]; write bf16 into hc_cat[:,512:1024]
__global__ __launch_bounds__(256)
void attn_ctx_kernel(const float* __restrict__ aw,
                     const bf16_t* __restrict__ enc_out,
                     bf16_t* __restrict__ dst, int ldd) {
    int b = blockIdx.x, e = threadIdx.x;        // e and e+256
    const bf16_t* p = enc_out + (size_t)b * SLEN * (2 * ENCH);
    const float* w = aw + b * SLEN;
    float a0 = 0.f, a1 = 0.f;
    for (int s = 0; s < SLEN; ++s) {
        float ws = w[s];
        a0 += ws * (float)p[s * (2 * ENCH) + e];
        a1 += ws * (float)p[s * (2 * ENCH) + e + 256];
    }
    dst[(size_t)b * ldd + e]       = (bf16_t)a0;
    dst[(size_t)b * ldd + e + 256] = (bf16_t)a1;
}

// feed = tanh(pre); store bf16 into x-concat feed slot (cols 16..527)
__global__ void feed_tanh_kernel(const float* __restrict__ pre,
                                 bf16_t* __restrict__ xdst, int ldx) {
    int idx = blockIdx.x * blockDim.x + threadIdx.x;   // B*512
    if (idx >= BSZ * DECH) return;
    int b = idx >> 9, j = idx & 511;
    xdst[(size_t)b * ldx + j] = (bf16_t)tanhf(pre[idx]);
}

// out[b,t,j] = act( feed[b,:].W_out[j,:] + b_out[j] ); tanh for j<3, relu j==3
__global__ void out_proj_kernel(const bf16_t* __restrict__ feed, int ldf,
                                const float* __restrict__ Wout,
                                const float* __restrict__ bout,
                                float* __restrict__ out, int t) {
    int idx = blockIdx.x * blockDim.x + threadIdx.x;   // B*DOF
    if (idx >= BSZ * DOF_) return;
    int b = idx >> 2, j = idx & 3;
    const bf16_t* f = feed + (size_t)b * ldf;
    const float* w = Wout + j * DECH;
    float acc = bout[j];
    for (int d = 0; d < DECH; ++d) acc += (float)f[d] * w[d];
    float r = (j == DOF_ - 1) ? fmaxf(acc, 0.f) : tanhf(acc);
    out[((size_t)b * TLEN + t) * DOF_ + j] = r;
}

// ---------------------------------------------------------------------------
// Host orchestration
// ---------------------------------------------------------------------------
extern "C" void kernel_launch(void* const* d_in, const int* in_sizes, int n_in,
                              void* d_out, int out_size, void* d_ws, size_t ws_size,
                              hipStream_t stream) {
    (void)in_sizes; (void)n_in; (void)out_size; (void)ws_size;

    const int*   in_seq    = (const int*)d_in[0];
    const float* tgt       = (const float*)d_in[1];
    /* d_in[2] lengths unused (mask derives from in_seq==0) */
    const float* embedding = (const float*)d_in[3];
    const float* Wih_f = (const float*)d_in[4];
    const float* Whh_f = (const float*)d_in[5];
    const float* b_f   = (const float*)d_in[6];
    const float* Wih_b = (const float*)d_in[7];
    const float* Whh_b = (const float*)d_in[8];
    const float* b_b   = (const float*)d_in[9];
    const float* dWih  = (const float*)d_in[10];
    const float* dWhh  = (const float*)d_in[11];
    const float* dB    = (const float*)d_in[12];
    const float* Wsrc  = (const float*)d_in[13];
    const float* Wctx  = (const float*)d_in[14];
    const float* Wout  = (const float*)d_in[15];
    const float* bout  = (const float*)d_in[16];

    float* out  = (float*)d_out;                  // [B,T,DOF]
    float* attn = out + (size_t)BSZ * TLEN * DOF_; // [B,T,S]

    // ---- workspace carve-up (256B aligned) ----
    char* wsb = (char*)d_ws;
    size_t o = 0;
    auto carve = [&](size_t bytes) -> char* {
        char* p = wsb + o;
        o += (bytes + 255) & ~(size_t)255;
        return p;
    };
    bf16_t* emb_bf  = (bf16_t*)carve((size_t)SLEN * BSZ * EMB * 2);
    bf16_t* win_bf  = (bf16_t*)carve((size_t)TLEN * BSZ * 16 * 2);
    bf16_t* Wcat_f  = (bf16_t*)carve((size_t)1024 * 512 * 2);
    bf16_t* Wcat_b  = (bf16_t*)carve((size_t)1024 * 512 * 2);
    bf16_t* Wdec    = (bf16_t*)carve((size_t)2048 * 1056 * 2);
    bf16_t* Wsrc_bf = (bf16_t*)carve((size_t)512 * 512 * 2);
    bf16_t* Wctx_bf = (bf16_t*)carve((size_t)512 * 1024 * 2);
    bf16_t* enc_out = (bf16_t*)carve((size_t)BSZ * SLEN * 512 * 2);
    float*  proj    = (float*) carve((size_t)BSZ * SLEN * 512 * 4);
    float*  gates_f = (float*) carve((size_t)BSZ * 1024 * 4);
    float*  gates_b = (float*) carve((size_t)BSZ * 1024 * 4);
    float*  gates_d = (float*) carve((size_t)BSZ * 2048 * 4);
    bf16_t* xcat_f  = (bf16_t*)carve((size_t)BSZ * 512 * 2);
    bf16_t* xcat_b  = (bf16_t*)carve((size_t)BSZ * 512 * 2);
    bf16_t* xcat_d  = (bf16_t*)carve((size_t)BSZ * 1056 * 2);
    float*  c_f     = (float*) carve((size_t)BSZ * ENCH * 4);
    float*  c_b     = (float*) carve((size_t)BSZ * ENCH * 4);
    float*  c_d     = (float*) carve((size_t)BSZ * DECH * 4);
    float*  h_d     = (float*) carve((size_t)BSZ * DECH * 4);
    bf16_t* hc_cat  = (bf16_t*)carve((size_t)BSZ * 1024 * 2);
    float*  aw      = (float*) carve((size_t)BSZ * SLEN * 4);
    float*  feedpre = (float*) carve((size_t)BSZ * DECH * 4);

    // ---- zero recurrent state (ws is poisoned; must re-init every call) ----
    hipMemsetAsync(xcat_f, 0, (size_t)BSZ * 512 * 2, stream);
    hipMemsetAsync(xcat_b, 0, (size_t)BSZ * 512 * 2, stream);
    hipMemsetAsync(xcat_d, 0, (size_t)BSZ * 1056 * 2, stream);  // pad cols stay 0
    hipMemsetAsync(c_f, 0, (size_t)BSZ * ENCH * 4, stream);
    hipMemsetAsync(c_b, 0, (size_t)BSZ * ENCH * 4, stream);
    hipMemsetAsync(c_d, 0, (size_t)BSZ * DECH * 4, stream);
    hipMemsetAsync(h_d, 0, (size_t)BSZ * DECH * 4, stream);
    hipMemsetAsync(hc_cat, 0, (size_t)BSZ * 1024 * 2, stream);

    // ---- one-time conversions / packs ----
    f32_to_bf16_kernel<<<(512 * 512 + 255) / 256, 256, 0, stream>>>(Wsrc, Wsrc_bf, 512 * 512);
    f32_to_bf16_kernel<<<(512 * 1024 + 255) / 256, 256, 0, stream>>>(Wctx, Wctx_bf, 512 * 1024);
    pack2_bf16_kernel<<<(1024 * 512 + 255) / 256, 256, 0, stream>>>(Wih_f, 256, Whh_f, 256, Wcat_f, 512, 1024);
    pack2_bf16_kernel<<<(1024 * 512 + 255) / 256, 256, 0, stream>>>(Wih_b, 256, Whh_b, 256, Wcat_b, 512, 1024);
    pack2_bf16_kernel<<<(2048 * 1056 + 255) / 256, 256, 0, stream>>>(dWih, 528, dWhh, 512, Wdec, 1056, 2048);
    embed_kernel<<<(SLEN * BSZ * EMB + 255) / 256, 256, 0, stream>>>(in_seq, embedding, emb_bf);
    win_kernel<<<(TLEN * BSZ * 16 + 255) / 256, 256, 0, stream>>>(tgt, win_bf);

    // ---- encoder: 256 steps, fwd + bwd interleaved ----
    const dim3 genc(1024 / 64, BSZ / 16);   // (16, 8)
    for (int t = 0; t < SLEN; ++t) {
        int tb = SLEN - 1 - t;
        enc_setx_kernel<<<(BSZ * EMB + 255) / 256, 256, 0, stream>>>(emb_bf, t, tb, xcat_f, xcat_b);
        wmma_gemm_bf16<<<genc, 128, 0, stream>>>(xcat_f, 512, Wcat_f, 512, b_f, gates_f, 1024, 512);
        wmma_gemm_bf16<<<genc, 128, 0, stream>>>(xcat_b, 512, Wcat_b, 512, b_b, gates_b, 1024, 512);
        lstm_cell_kernel<<<(BSZ * ENCH + 255) / 256, 256, 0, stream>>>(
            gates_f, c_f, xcat_f + EMB, 512,
            enc_out + (size_t)t * 512, (long)SLEN * 512, nullptr, BSZ, ENCH);
        lstm_cell_kernel<<<(BSZ * ENCH + 255) / 256, 256, 0, stream>>>(
            gates_b, c_b, xcat_b + EMB, 512,
            enc_out + (size_t)tb * 512 + ENCH, (long)SLEN * 512, nullptr, BSZ, ENCH);
    }

    // ---- attention source projection: proj[b,s,:] = enc_out[b,s,:] @ Wsrc^T ----
    {
        dim3 gp(512 / 64, (BSZ * SLEN) / 16);   // (8, 2048)
        wmma_gemm_bf16<<<gp, 128, 0, stream>>>(enc_out, 512, Wsrc_bf, 512, nullptr, proj, 512, 512);
    }

    // ---- decoder: 256 steps ----
    const dim3 gdec(2048 / 64, BSZ / 16);   // (32, 8)
    const dim3 gfeed(512 / 64, BSZ / 16);   // (8, 8)
    for (int t = 0; t < TLEN; ++t) {
        dec_setwin_kernel<<<(BSZ * 16 + 255) / 256, 256, 0, stream>>>(win_bf, t, xcat_d);
        wmma_gemm_bf16<<<gdec, 128, 0, stream>>>(xcat_d, 1056, Wdec, 1056, dB, gates_d, 2048, 1056);
        lstm_cell_kernel<<<(BSZ * DECH + 255) / 256, 256, 0, stream>>>(
            gates_d, c_d, xcat_d + 528, 1056, hc_cat, 1024, h_d, BSZ, DECH);
        attn_scores_kernel<<<BSZ, 256, 0, stream>>>(h_d, proj, in_seq, aw, attn, t);
        attn_ctx_kernel<<<BSZ, 256, 0, stream>>>(aw, enc_out, hc_cat + 512, 1024);
        wmma_gemm_bf16<<<gfeed, 128, 0, stream>>>(hc_cat, 1024, Wctx_bf, 1024, nullptr, feedpre, 512, 1024);
        feed_tanh_kernel<<<(BSZ * DECH + 255) / 256, 256, 0, stream>>>(feedpre, xcat_d + 16, 1056);
        out_proj_kernel<<<(BSZ * DOF_ + 255) / 256, 256, 0, stream>>>(xcat_d + 16, 1056, Wout, bout, out, t);
    }
}
